// TMSA_1563368096272
// MI455X (gfx1250) — compile-verified, matrix operand order checked by reading, not run on hardware
//
#include <hip/hip_runtime.h>
#include <hip/hip_bf16.h>

typedef __attribute__((ext_vector_type(16))) __bf16 v16bf;
typedef __attribute__((ext_vector_type(8)))  float  v8f;
typedef unsigned short u16;
typedef unsigned int   u32;

#define C_DIM   192
#define NHEADS  6
#define HD      32
#define NTOK    128
#define NWIN    1024
#define NWIN_B  512
#define NTOKENS 131072

// ---------- bf16 helpers ----------
__device__ __forceinline__ u16 f2bf(float f) {
  union { float f; u32 u; } v; v.f = f;
  u32 r = v.u + 0x7FFFu + ((v.u >> 16) & 1u);
  return (u16)(r >> 16);
}
__device__ __forceinline__ float bf2f(u16 h) {
  union { u32 u; float f; } v; v.u = ((u32)h) << 16;
  return v.f;
}
__device__ __forceinline__ v8f zero8() {
  v8f z = {0.f,0.f,0.f,0.f,0.f,0.f,0.f,0.f}; return z;
}

union Frag {
  u16 s[16];
  u32 d[8];
  v16bf b;
};

// ---------- CDNA5 async global->LDS staging (ASYNCcnt path) ----------
// LDS byte offset = low 32 bits of the generic shared-space address (aperture rule).
__device__ __forceinline__ void async_b128(u16* lds_dst, const u16* gsrc) {
  u32 loff = (u32)(size_t)lds_dst;
  asm volatile("global_load_async_to_lds_b128 %0, %1, off"
               :: "v"(loff), "v"(gsrc) : "memory");
}
__device__ __forceinline__ void async_fence() {
  asm volatile("s_wait_asynccnt 0x0" ::: "memory");
}
// copy nbytes (multiple of 16) from global to LDS cooperatively, async
__device__ __forceinline__ void stage_async(u16* lds, const u16* g, int n_u16) {
  const int chunks = n_u16 >> 3;     // 16B chunks
  for (int i = threadIdx.x; i < chunks; i += blockDim.x)
    async_b128(lds + i * 8, g + i * 8);
}

// A-matrix 16x32 bf16 fragment (also used for B = W^T when W is row-major over K):
// lane m = lane&15 holds row m; VGPR j packs K pair; K = (lane>>4)*8 + ((j&4)?16:0) + (j&3)*2
__device__ __forceinline__ v16bf load_frag_row(const u16* base, int ld, int row0, int col0) {
  const int lane = threadIdx.x & 31;
  const int m = lane & 15, hi = lane >> 4;
  Frag f;
  const u16* rp = base + (size_t)(row0 + m) * ld + col0 + hi * 8;
#pragma unroll
  for (int j = 0; j < 8; ++j) {
    const int k0 = ((j & 4) ? 16 : 0) + (j & 3) * 2;
    f.d[j] = *(const u32*)(rp + k0);
  }
  return f.b;
}

// B-matrix 32x16 where memory is row-major K x N (e.g. V: [k][32])
__device__ __forceinline__ v16bf load_frag_col(const u16* base, int ld, int k0r, int col0) {
  const int lane = threadIdx.x & 31;
  const int n = lane & 15, hi = lane >> 4;
  Frag f;
#pragma unroll
  for (int j = 0; j < 8; ++j) {
    const int k0 = ((j & 4) ? 16 : 0) + hi * 8 + (j & 3) * 2;
    f.s[2*j]   = base[(size_t)(k0r + k0    ) * ld + col0 + n];
    f.s[2*j+1] = base[(size_t)(k0r + k0 + 1) * ld + col0 + n];
  }
  return f.b;
}

__device__ __forceinline__ v8f wmma_bf(v16bf a, v16bf b, v8f c) {
  return __builtin_amdgcn_wmma_f32_16x16x32_bf16(false, a, false, b, (short)0, c, false, false);
}

// ---------- small precompute kernels ----------
__global__ void conv_bf16_kernel(const float* __restrict__ src, u16* __restrict__ dst, int n) {
  int i = blockIdx.x * blockDim.x + threadIdx.x;
  if (i < n) dst[i] = f2bf(src[i]);
}

__global__ void pos_kernel(u16* __restrict__ pos) {
  int i = blockIdx.x * blockDim.x + threadIdx.x;
  if (i >= NTOK * C_DIM) return;
  int n = i / C_DIM, c = i % C_DIM;
  int p = n & 63, yi = p >> 3, xj = p & 7;
  const float scale = 6.283185307179586f;
  float v = (c < 96) ? ((float)(yi + 1) / (8.0f + 1e-6f) * scale)
                     : ((float)(xj + 1) / (8.0f + 1e-6f) * scale);
  int f = (c < 96) ? c : c - 96;
  float e = (float)(2 * (f / 2)) / 96.0f;
  float a = v / powf(10000.0f, e);
  pos[i] = f2bf((f & 1) ? cosf(a) : sinf(a));
}

__global__ void rpb_kernel(const float* __restrict__ tbl, float* __restrict__ rpbful) {
  int i = blockIdx.x * blockDim.x + threadIdx.x;
  if (i >= NTOK * NTOK) return;
  int a = i / NTOK, b = i % NTOK;
  int da = a >> 6, ha = (a >> 3) & 7, wa = a & 7;
  int dB = b >> 6, hB = (b >> 3) & 7, wB = b & 7;
  int idx = (da - dB + 1) * 225 + (ha - hB + 7) * 15 + (wa - wB + 7);
#pragma unroll
  for (int h = 0; h < NHEADS; ++h)
    rpbful[((size_t)h * NTOK + a) * NTOK + b] = tbl[idx * NHEADS + h];
}

// ---------- LN1 + shift + window partition -> XW bf16 [win][128][192] ----------
__global__ void ln1_kernel(const float* __restrict__ x, const float* __restrict__ w,
                           const float* __restrict__ b, u16* __restrict__ xw) {
  int gid = blockIdx.x;               // win*128 + n
  int win = gid >> 7, n = gid & 127;
  int bN = win >> 9, rem = win & 511;
  int db = rem >> 8, hb = (rem >> 4) & 15, wb = rem & 15;
  int dd = n >> 6, hh = (n >> 3) & 7, ww2 = n & 7;
  int d0 = (db * 2 + dd + 1) & 3;
  int h0 = (hb * 8 + hh + 4) & 127;
  int w0 = (wb * 8 + ww2 + 4) & 127;
  const float* row = x + ((((size_t)bN * 4 + d0) * 128 + h0) * 128 + w0) * C_DIM;
  int c = threadIdx.x;
  float v = row[c];
  __shared__ float red[C_DIM];
  red[c] = v; __syncthreads();
  for (int s = 128; s > 0; s >>= 1) {
    if (c < s && c + s < C_DIM) red[c] += red[c + s];
    __syncthreads();
  }
  float mu = red[0] / (float)C_DIM;
  __syncthreads();
  float dv = v - mu;
  red[c] = dv * dv; __syncthreads();
  for (int s = 128; s > 0; s >>= 1) {
    if (c < s && c + s < C_DIM) red[c] += red[c + s];
    __syncthreads();
  }
  float var = red[0] / (float)C_DIM;
  xw[(size_t)gid * C_DIM + c] = f2bf(dv * rsqrtf(var + 1e-5f) * w[c] + b[c]);
}

// ---------- QKV GEMM: [128,192] @ [576,192]^T -> q,k,v bf16 [3][win][head][tok][32] ----------
__global__ void qkv_gemm_kernel(const u16* __restrict__ xw, const u16* __restrict__ pos,
                                const u16* __restrict__ wbf, const float* __restrict__ bias,
                                u16* __restrict__ qkv, int addpos) {
  int win = blockIdx.x;
  extern __shared__ __align__(16) char smem_raw[];
  u16* Xs = (u16*)smem_raw;           // 128*192
  const u16* src = xw + (size_t)win * NTOK * C_DIM;
  __builtin_prefetch(wbf + (size_t)threadIdx.x * 432, 0, 1);   // warm GL2 with weights
  if (addpos) {
    for (int i = threadIdx.x; i < NTOK * C_DIM; i += blockDim.x)
      Xs[i] = f2bf(bf2f(src[i]) + bf2f(pos[i]));
  } else {
    stage_async(Xs, src, NTOK * C_DIM);
    async_fence();
  }
  __syncthreads();
  const int wave = threadIdx.x >> 5, lane = threadIdx.x & 31;
  const int ln = lane & 15, lh = lane >> 4;
  for (int t = wave; t < 8 * 36; t += 8) {
    int mt = t / 36, nt = t % 36;
    v8f acc = zero8();
#pragma unroll
    for (int kk = 0; kk < 6; ++kk) {
      v16bf a = load_frag_row(Xs, C_DIM, mt * 16, kk * 32);
      v16bf bb = load_frag_row(wbf, C_DIM, nt * 16, kk * 32);
      acc = wmma_bf(a, bb, acc);
    }
    int o = nt * 16 + ln;
    int s = o / C_DIM, remc = o % C_DIM, head = remc / HD, wi = remc % HD;
    float bs = bias[o];
    float scale = (s == 0) ? 0.17677669529663689f : 1.0f;   // q * head_dim^-0.5
    u16* dst = qkv + (((size_t)s * NWIN + win) * NHEADS + head) * (NTOK * HD) + wi;
#pragma unroll
    for (int r = 0; r < 8; ++r) {
      int tok = mt * 16 + lh * 8 + r;
      dst[(size_t)tok * HD] = f2bf((acc[r] + bs) * scale);
    }
  }
}

// ---------- self attention per (win, head): 128x128 scores + rpb + mask, softmax, @V ----------
__global__ void attn_self_kernel(const u16* __restrict__ qkv, const float* __restrict__ rpbful,
                                 const float* __restrict__ mask, u16* __restrict__ xout) {
  const int win = blockIdx.x / NHEADS;
  const int head = blockIdx.x % NHEADS;
  const int wm = win % NWIN_B;
  extern __shared__ __align__(16) char smem_raw[];
  u16* S  = (u16*)smem_raw;           // 128*128 bf16
  u16* Qs = S + NTOK * NTOK;          // 128*32
  u16* Ks = Qs + NTOK * HD;
  u16* Vs = Ks + NTOK * HD;
  const size_t hb = ((size_t)win * NHEADS + head) * (NTOK * HD);
  const size_t stride = (size_t)NWIN * NHEADS * NTOK * HD;
  stage_async(Qs, qkv + hb, NTOK * HD);
  stage_async(Ks, qkv + stride + hb, NTOK * HD);
  stage_async(Vs, qkv + 2 * stride + hb, NTOK * HD);
  async_fence();
  __syncthreads();
  const int wave = threadIdx.x >> 5, lane = threadIdx.x & 31;
  const int ln = lane & 15, lh = lane >> 4;
  // S = (Q*scale) K^T, + rpb + mask
  for (int t = wave; t < 64; t += 8) {
    int mt = t >> 3, nt = t & 7;
    v16bf a = load_frag_row(Qs, HD, mt * 16, 0);
    v16bf bb = load_frag_row(Ks, HD, nt * 16, 0);   // B[k][n] = K[n][k]
    v8f acc = zero8();
    acc = wmma_bf(a, bb, acc);
#pragma unroll
    for (int r = 0; r < 8; ++r) {
      int i = mt * 16 + lh * 8 + r;
      int j = nt * 16 + ln;
      float sv = acc[r] + rpbful[((size_t)head * NTOK + i) * NTOK + j]
                        + mask[((size_t)wm * NTOK + i) * NTOK + j];
      S[i * NTOK + j] = f2bf(sv);
    }
  }
  __syncthreads();
  if (threadIdx.x < NTOK) {
    int i = threadIdx.x;
    u16* row = S + i * NTOK;
    float mx = -3.0e38f;
    for (int j = 0; j < NTOK; ++j) mx = fmaxf(mx, bf2f(row[j]));
    float sum = 0.f;
    for (int j = 0; j < NTOK; ++j) { float e = __expf(bf2f(row[j]) - mx); row[j] = f2bf(e); sum += e; }
    float inv = 1.0f / sum;
    for (int j = 0; j < NTOK; ++j) row[j] = f2bf(bf2f(row[j]) * inv);
  }
  __syncthreads();
  // O = P @ V -> xout channels [192 + head*32 .. )
  for (int t = wave; t < 16; t += 8) {
    int mt = t >> 1, nt = t & 1;
    v8f acc = zero8();
#pragma unroll
    for (int kk = 0; kk < 4; ++kk) {
      v16bf a = load_frag_row(S, NTOK, mt * 16, kk * 32);
      v16bf bb = load_frag_col(Vs, HD, kk * 32, nt * 16);
      acc = wmma_bf(a, bb, acc);
    }
#pragma unroll
    for (int r = 0; r < 8; ++r) {
      int tok = mt * 16 + lh * 8 + r;
      int ch = C_DIM + head * HD + nt * 16 + ln;
      xout[((size_t)win * NTOK + tok) * 384 + ch] = f2bf(acc[r]);
    }
  }
}

// ---------- mutual attention per (win, head, half): 64x64 ----------
__global__ void attn_mut_kernel(const u16* __restrict__ qkv, const float* __restrict__ mask,
                                u16* __restrict__ xout) {
  int id = blockIdx.x;
  const int half = id & 1; id >>= 1;
  const int head = id % NHEADS;
  const int win = id / NHEADS;
  const int wm = win % NWIN_B;
  extern __shared__ __align__(16) char smem_raw[];
  u16* S  = (u16*)smem_raw;          // 64*64
  u16* Qs = S + 64 * 64;             // 64*32
  u16* Ks = Qs + 64 * HD;
  u16* Vs = Ks + 64 * HD;
  const size_t hb = ((size_t)win * NHEADS + head) * (NTOK * HD);
  const size_t stride = (size_t)NWIN * NHEADS * NTOK * HD;
  const u16* qg = qkv + hb + (size_t)((half == 0) ? 64 : 0) * HD;  // x1 uses q2
  const u16* kg = qkv + stride + hb + (size_t)((half == 0) ? 0 : 64) * HD;
  const u16* vg = qkv + 2 * stride + hb + (size_t)((half == 0) ? 0 : 64) * HD;
  stage_async(Qs, qg, 64 * HD);
  stage_async(Ks, kg, 64 * HD);
  stage_async(Vs, vg, 64 * HD);
  async_fence();
  __syncthreads();
  const int wave = threadIdx.x >> 5, lane = threadIdx.x & 31;
  const int ln = lane & 15, lh = lane >> 4;
  for (int t = wave; t < 16; t += 4) {
    int mt = t >> 2, nt = t & 3;
    v16bf a = load_frag_row(Qs, HD, mt * 16, 0);
    v16bf bb = load_frag_row(Ks, HD, nt * 16, 0);
    v8f acc = zero8();
    acc = wmma_bf(a, bb, acc);
#pragma unroll
    for (int r = 0; r < 8; ++r) {
      int i = mt * 16 + lh * 8 + r;
      int j = nt * 16 + ln;
      float sv = acc[r] + mask[((size_t)wm * NTOK + i) * NTOK + j];  // m2 = mask[:, :64, :64]
      S[i * 64 + j] = f2bf(sv);
    }
  }
  __syncthreads();
  if (threadIdx.x < 64) {
    int i = threadIdx.x;
    u16* row = S + i * 64;
    float mx = -3.0e38f;
    for (int j = 0; j < 64; ++j) mx = fmaxf(mx, bf2f(row[j]));
    float sum = 0.f;
    for (int j = 0; j < 64; ++j) { float e = __expf(bf2f(row[j]) - mx); row[j] = f2bf(e); sum += e; }
    float inv = 1.0f / sum;
    for (int j = 0; j < 64; ++j) row[j] = f2bf(bf2f(row[j]) * inv);
  }
  __syncthreads();
  for (int t = wave; t < 8; t += 4) {
    int mt = t >> 1, nt = t & 1;
    v8f acc = zero8();
#pragma unroll
    for (int kk = 0; kk < 2; ++kk) {
      v16bf a = load_frag_row(S, 64, mt * 16, kk * 32);
      v16bf bb = load_frag_col(Vs, HD, kk * 32, nt * 16);
      acc = wmma_bf(a, bb, acc);
    }
#pragma unroll
    for (int r = 0; r < 8; ++r) {
      int tok = half * 64 + mt * 16 + lh * 8 + r;
      int ch = head * HD + nt * 16 + ln;
      xout[((size_t)win * NTOK + tok) * 384 + ch] = f2bf(acc[r]);
    }
  }
}

// ---------- proj GEMM [128,384]@[192,384]^T + window-reverse + roll + residual -> XRES f32 ----------
__global__ void proj_kernel(const u16* __restrict__ xout, const u16* __restrict__ wproj,
                            const float* __restrict__ pb, const float* __restrict__ x0,
                            float* __restrict__ xres) {
  int win = blockIdx.x;
  extern __shared__ __align__(16) char smem_raw[];
  u16* Xs = (u16*)smem_raw;          // 128*384
  __builtin_prefetch(wproj + (size_t)threadIdx.x * 288, 0, 1);
  stage_async(Xs, xout + (size_t)win * NTOK * 384, NTOK * 384);
  async_fence();
  __syncthreads();
  int bN = win >> 9, rem = win & 511;
  int db = rem >> 8, hb = (rem >> 4) & 15, wb = rem & 15;
  const int wave = threadIdx.x >> 5, lane = threadIdx.x & 31;
  const int ln = lane & 15, lh = lane >> 4;
  for (int t = wave; t < 96; t += 8) {
    int mt = t / 12, nt = t % 12;
    v8f acc = zero8();
#pragma unroll
    for (int kk = 0; kk < 12; ++kk) {
      v16bf a = load_frag_row(Xs, 384, mt * 16, kk * 32);
      v16bf bb = load_frag_row(wproj, 384, nt * 16, kk * 32);
      acc = wmma_bf(a, bb, acc);
    }
    int o = nt * 16 + ln;
    float bs = pb[o];
#pragma unroll
    for (int r = 0; r < 8; ++r) {
      int tok = mt * 16 + lh * 8 + r;
      int dd = tok >> 6, hh = (tok >> 3) & 7, ww2 = tok & 7;
      int d0 = (db * 2 + dd + 1) & 3;
      int h0 = (hb * 8 + hh + 4) & 127;
      int w0 = (wb * 8 + ww2 + 4) & 127;
      size_t idx = ((((size_t)bN * 4 + d0) * 128 + h0) * 128 + w0) * C_DIM + o;
      xres[idx] = x0[idx] + acc[r] + bs;
    }
  }
}

// ---------- LN2 -> H2 bf16 ----------
__global__ void ln2_kernel(const float* __restrict__ xres, const float* __restrict__ w,
                           const float* __restrict__ b, u16* __restrict__ h2) {
  size_t tid = blockIdx.x;
  const float* row = xres + tid * C_DIM;
  int c = threadIdx.x;
  float v = row[c];
  __shared__ float red[C_DIM];
  red[c] = v; __syncthreads();
  for (int s = 128; s > 0; s >>= 1) {
    if (c < s && c + s < C_DIM) red[c] += red[c + s];
    __syncthreads();
  }
  float mu = red[0] / (float)C_DIM;
  __syncthreads();
  float dv = v - mu;
  red[c] = dv * dv; __syncthreads();
  for (int s = 128; s > 0; s >>= 1) {
    if (c < s && c + s < C_DIM) red[c] += red[c + s];
    __syncthreads();
  }
  float var = red[0] / (float)C_DIM;
  h2[tid * C_DIM + c] = f2bf(dv * rsqrtf(var + 1e-5f) * w[c] + b[c]);
}

// ---------- gated MLP: g = gelu(h2@W11^T+b11) * (h2@W12^T+b12) bf16 [N,384] ----------
__global__ void glu_kernel(const u16* __restrict__ h2, const u16* __restrict__ w11,
                           const float* __restrict__ b11, const u16* __restrict__ w12,
                           const float* __restrict__ b12, u16* __restrict__ g) {
  int rb = blockIdx.x;               // block of 128 tokens
  extern __shared__ __align__(16) char smem_raw[];
  u16* Xs = (u16*)smem_raw;          // 128*192
  __builtin_prefetch(w11 + (size_t)threadIdx.x * 288, 0, 1);
  __builtin_prefetch(w12 + (size_t)threadIdx.x * 288, 0, 1);
  stage_async(Xs, h2 + (size_t)rb * 128 * C_DIM, 128 * C_DIM);
  async_fence();
  __syncthreads();
  const int wave = threadIdx.x >> 5, lane = threadIdx.x & 31;
  const int ln = lane & 15, lh = lane >> 4;
  for (int t = wave; t < 8 * 24; t += 8) {
    int mt = t / 24, nt = t % 24;
    v8f a1 = zero8(), a2 = zero8();
#pragma unroll
    for (int kk = 0; kk < 6; ++kk) {
      v16bf a = load_frag_row(Xs, C_DIM, mt * 16, kk * 32);
      a1 = wmma_bf(a, load_frag_row(w11, C_DIM, nt * 16, kk * 32), a1);
      a2 = wmma_bf(a, load_frag_row(w12, C_DIM, nt * 16, kk * 32), a2);
    }
    int o = nt * 16 + ln;
    float c1 = b11[o], c2 = b12[o];
#pragma unroll
    for (int r = 0; r < 8; ++r) {
      int row = rb * 128 + mt * 16 + lh * 8 + r;
      float u = a1[r] + c1;
      float vv = a2[r] + c2;
      float ge = 0.5f * u * (1.0f + erff(u * 0.70710678118654752f));
      g[(size_t)row * 384 + o] = f2bf(ge * vv);
    }
  }
}

// ---------- final: out = xres + g@W2^T + b2 (f32) ----------
__global__ void out_kernel(const u16* __restrict__ g, const u16* __restrict__ w2,
                           const float* __restrict__ b2, const float* __restrict__ xres,
                           float* __restrict__ out) {
  int rb = blockIdx.x;
  extern __shared__ __align__(16) char smem_raw[];
  u16* Gs = (u16*)smem_raw;          // 128*384
  __builtin_prefetch(w2 + (size_t)threadIdx.x * 288, 0, 1);
  stage_async(Gs, g + (size_t)rb * 128 * 384, 128 * 384);
  async_fence();
  __syncthreads();
  const int wave = threadIdx.x >> 5, lane = threadIdx.x & 31;
  const int ln = lane & 15, lh = lane >> 4;
  for (int t = wave; t < 96; t += 8) {
    int mt = t / 12, nt = t % 12;
    v8f acc = zero8();
#pragma unroll
    for (int kk = 0; kk < 12; ++kk) {
      v16bf a = load_frag_row(Gs, 384, mt * 16, kk * 32);
      v16bf bb = load_frag_row(w2, 384, nt * 16, kk * 32);
      acc = wmma_bf(a, bb, acc);
    }
    int o = nt * 16 + ln;
    float bs = b2[o];
#pragma unroll
    for (int r = 0; r < 8; ++r) {
      int row = rb * 128 + mt * 16 + lh * 8 + r;
      size_t idx = (size_t)row * C_DIM + o;
      out[idx] = xres[idx] + acc[r] + bs;
    }
  }
}

// ---------- launch ----------
extern "C" void kernel_launch(void* const* d_in, const int* in_sizes, int n_in,
                              void* d_out, int out_size, void* d_ws, size_t ws_size,
                              hipStream_t stream) {
  (void)in_sizes; (void)n_in; (void)out_size; (void)ws_size;
  const float* x     = (const float*)d_in[0];
  const float* mask  = (const float*)d_in[1];
  const float* n1w   = (const float*)d_in[2];
  const float* n1b   = (const float*)d_in[3];
  const float* wqs   = (const float*)d_in[4];
  const float* bqs   = (const float*)d_in[5];
  const float* wqm   = (const float*)d_in[6];
  const float* bqm   = (const float*)d_in[7];
  const float* rpbt  = (const float*)d_in[8];
  const float* wpr   = (const float*)d_in[9];
  const float* bpr   = (const float*)d_in[10];
  const float* n2w   = (const float*)d_in[11];
  const float* n2b   = (const float*)d_in[12];
  const float* w11   = (const float*)d_in[13];
  const float* b11   = (const float*)d_in[14];
  const float* w12   = (const float*)d_in[15];
  const float* b12   = (const float*)d_in[16];
  const float* w2    = (const float*)d_in[17];
  const float* b2    = (const float*)d_in[18];
  float* out = (float*)d_out;

  char* ws = (char*)d_ws;
  // small region
  const size_t OFF_WQS  = 0;                         // 110592 bf16
  const size_t OFF_WQM  = 221184;
  const size_t OFF_WPR  = 442368;                    // 73728 bf16
  const size_t OFF_W11  = 589824;
  const size_t OFF_W12  = 737280;
  const size_t OFF_W2   = 884736;
  const size_t OFF_POS  = 1032192;                   // 24576 bf16
  const size_t OFF_RPB  = 1081344;                   // 98304 f32
  const size_t BASE     = 2097152;
  // big buffers (with lifetime reuse)
  const size_t SZ_XW   = (size_t)NWIN * NTOK * C_DIM * 2;           // 50.3 MB
  const size_t SZ_QKV  = 3 * (size_t)NWIN * NHEADS * NTOK * HD * 2; // 151 MB
  const size_t OFF_XW   = BASE;
  const size_t OFF_QKV  = OFF_XW + SZ_XW;
  const size_t OFF_XOUT = OFF_QKV + SZ_QKV;
  const size_t OFF_XRES = OFF_QKV;    // f32, reuses dead QKV region
  const size_t OFF_H2   = OFF_XW;     // reuses dead XW region
  const size_t OFF_G    = OFF_XOUT;   // reuses dead XOUT region

  u16* WQS  = (u16*)(ws + OFF_WQS);
  u16* WQM  = (u16*)(ws + OFF_WQM);
  u16* WPR  = (u16*)(ws + OFF_WPR);
  u16* W11  = (u16*)(ws + OFF_W11);
  u16* W12  = (u16*)(ws + OFF_W12);
  u16* W2   = (u16*)(ws + OFF_W2);
  u16* POS  = (u16*)(ws + OFF_POS);
  float* RPB = (float*)(ws + OFF_RPB);
  u16* XW   = (u16*)(ws + OFF_XW);
  u16* QKV  = (u16*)(ws + OFF_QKV);
  u16* XOUT = (u16*)(ws + OFF_XOUT);
  float* XRES = (float*)(ws + OFF_XRES);
  u16* H2   = (u16*)(ws + OFF_H2);
  u16* G    = (u16*)(ws + OFF_G);

  // weight conversion + tables
  conv_bf16_kernel<<<(110592 + 255) / 256, 256, 0, stream>>>(wqs, WQS, 110592);
  conv_bf16_kernel<<<(110592 + 255) / 256, 256, 0, stream>>>(wqm, WQM, 110592);
  conv_bf16_kernel<<<(73728 + 255) / 256, 256, 0, stream>>>(wpr, WPR, 73728);
  conv_bf16_kernel<<<(73728 + 255) / 256, 256, 0, stream>>>(w11, W11, 73728);
  conv_bf16_kernel<<<(73728 + 255) / 256, 256, 0, stream>>>(w12, W12, 73728);
  conv_bf16_kernel<<<(73728 + 255) / 256, 256, 0, stream>>>(w2, W2, 73728);
  pos_kernel<<<(NTOK * C_DIM + 255) / 256, 256, 0, stream>>>(POS);
  rpb_kernel<<<(NTOK * NTOK + 255) / 256, 256, 0, stream>>>(rpbt, RPB);

  // LN1 + shift + partition
  ln1_kernel<<<NTOKENS, C_DIM, 0, stream>>>(x, n1w, n1b, XW);

  // self QKV -> self attention (QKV region then reused by mutual QKV)
  qkv_gemm_kernel<<<NWIN, 256, 128 * C_DIM * 2, stream>>>(XW, POS, WQS, bqs, QKV, 0);
  attn_self_kernel<<<NWIN * NHEADS, 256, (NTOK * NTOK + 3 * NTOK * HD) * 2, stream>>>(QKV, RPB, mask, XOUT);
  qkv_gemm_kernel<<<NWIN, 256, 128 * C_DIM * 2, stream>>>(XW, POS, WQM, bqm, QKV, 1);
  attn_mut_kernel<<<NWIN * NHEADS * 2, 128, (64 * 64 + 3 * 64 * HD) * 2, stream>>>(QKV, mask, XOUT);

  // proj + reverse + residual
  proj_kernel<<<NWIN, 256, 128 * 384 * 2, stream>>>(XOUT, WPR, bpr, x, XRES);

  // LN2 + gated MLP + residual
  ln2_kernel<<<NTOKENS, C_DIM, 0, stream>>>(XRES, n2w, n2b, H2);
  glu_kernel<<<NTOKENS / 128, 256, 128 * C_DIM * 2, stream>>>(H2, W11, b11, W12, b12, G);
  out_kernel<<<NTOKENS / 128, 256, 128 * 384 * 2, stream>>>(G, W2, b2, XRES, out);
}
